// ProbAttention_23502061044325
// MI455X (gfx1250) — matrix operand order repaired and checked by reading, not compiled, not run on hardware
//
#include <hip/hip_runtime.h>
#include <hip/hip_bf16.h>
#include <hip/hip_fp16.h>

typedef __attribute__((ext_vector_type(16))) _Float16 v16h;
typedef __attribute__((ext_vector_type(8)))  _Float16 v8h;
typedef __attribute__((ext_vector_type(8)))  float    v8f;
typedef __attribute__((ext_vector_type(4)))  float    v4f;

#define B_    4
#define L_    2048
#define H_    8
#define D_    64
#define SK_   40      // sampled keys per query
#define U_    40      // top-u queries kept
#define UPAD_ 48      // padded to 3 tiles of 16
#define QT_   3       // query tiles of 16
#define SCALE_ 0.125f // 1/sqrt(64)
#define NEGBIG (-3.0e38f)

// ---------------------------------------------------------------------------
// Kernel 1: M[bh][l] = max_s(Q.Ks) - sum_s(Q.Ks)/L   (one wave32 per query)
// ---------------------------------------------------------------------------
__global__ void probM_kernel(const float* __restrict__ q,
                             const float* __restrict__ k,
                             const long long* __restrict__ idx,
                             float* __restrict__ Mout) {
  int wid  = blockIdx.x * (blockDim.x >> 5) + (threadIdx.x >> 5);
  int lane = threadIdx.x & 31;
  if (wid >= B_ * H_ * L_) return;
  int l  = wid % L_;
  int bh = wid / L_;
  int h  = bh % H_;
  int b  = bh / H_;
  const float* qrow = q + (((size_t)(b * L_ + l) * H_ + h) * D_);
  float q0 = qrow[lane];
  float q1 = qrow[lane + 32];
  float mx = NEGBIG, sm = 0.0f;
  for (int s = 0; s < SK_; ++s) {
    long long ks = idx[(long long)l * SK_ + s];
    const float* krow = k + (((size_t)(b * L_ + (int)ks) * H_ + h) * D_);
    float p = q0 * krow[lane] + q1 * krow[lane + 32];
    for (int off = 16; off > 0; off >>= 1) p += __shfl_xor(p, off, 32);
    mx = fmaxf(mx, p);
    sm += p;
  }
  if (lane == 0) Mout[(size_t)bh * L_ + l] = mx - sm * (1.0f / (float)L_);
}

// ---------------------------------------------------------------------------
// Kernel 2: per (b,h) iterative top-40 of M (tie -> lower index, like top_k),
// then gather selected Q rows, pre-scaled by 1/sqrt(D), as f16, zero-padded.
// ---------------------------------------------------------------------------
__global__ void topk_gather_kernel(const float* __restrict__ Min,
                                   const float* __restrict__ q,
                                   int* __restrict__ Mtop,
                                   _Float16* __restrict__ Qr) {
  __shared__ float mv[L_];
  __shared__ float redv[256];
  __shared__ int   redi[256];
  __shared__ int   topIdx[U_];
  int bh  = blockIdx.x;
  int tid = threadIdx.x;
  int h = bh % H_, b = bh / H_;

  for (int i = tid; i < L_; i += 256) mv[i] = Min[(size_t)bh * L_ + i];
  __syncthreads();

  for (int r = 0; r < U_; ++r) {
    float bv = NEGBIG; int bi = L_;
    for (int i = tid; i < L_; i += 256) {
      float vv = mv[i];
      if (vv > bv || (vv == bv && i < bi)) { bv = vv; bi = i; }
    }
    redv[tid] = bv; redi[tid] = bi;
    __syncthreads();
    for (int s = 128; s > 0; s >>= 1) {
      if (tid < s) {
        float ov = redv[tid + s]; int oi = redi[tid + s];
        if (ov > redv[tid] || (ov == redv[tid] && oi < redi[tid])) {
          redv[tid] = ov; redi[tid] = oi;
        }
      }
      __syncthreads();
    }
    if (tid == 0) {
      int sel = redi[0];
      topIdx[r] = sel;
      Mtop[bh * U_ + r] = sel;
      mv[sel] = NEGBIG;
    }
    __syncthreads();
  }

  for (int i = tid; i < UPAD_ * D_; i += 256) {
    int r = i / D_, d = i % D_;
    float val = 0.0f;
    if (r < U_) {
      int l = topIdx[r];
      val = q[((size_t)(b * L_ + l) * H_ + h) * D_ + d] * SCALE_;
    }
    Qr[((size_t)bh * UPAD_ + r) * D_ + d] = (_Float16)val;
  }
}

// ---------------------------------------------------------------------------
// Kernel 3: flash attention for selected queries, WMMA f16->f32.
// One wave32 per (b,h, 16-query tile). Streams 64 blocks of 32 keys.
// K/V staged through LDS: coalesced b128 global loads, b128 LDS loads for
// the S-GEMM B-fragments. Softmax butterflies are batched 8-wide so the
// ds_bpermute steps issue in parallel (one dscnt wait per step, not per row).
// ---------------------------------------------------------------------------
__global__ __launch_bounds__(32)
void probattn_kernel(const _Float16* __restrict__ Qr,
                     const float* __restrict__ kmat,
                     const float* __restrict__ vmat,
                     const int* __restrict__ amask,
                     float* __restrict__ out) {
  __shared__ _Float16 kbuf[32][D_];   // 4KB: 32-key block of K, f16 row-major
  __shared__ _Float16 vbuf[32][D_];   // 4KB: 32-key block of V
  __shared__ _Float16 pbuf[16][32];   // 1KB: P tile staged for A-frag reload

  int blk  = blockIdx.x;
  int qt   = blk % QT_;
  int bh   = blk / QT_;
  int h    = bh % H_, b = bh / H_;
  int lane = threadIdx.x;
  int half = lane >> 4;     // 0/1
  int nn   = lane & 15;     // N column (B/C frags) and M row (A frags)

  // A fragments of Qr tile (16x64 -> two 16x32). Per-lane data is two
  // contiguous 8-element f16 runs per fragment -> vector loads.
  const _Float16* qbase = Qr + ((size_t)bh * UPAD_ + qt * 16 + nn) * D_;
  v16h a_lo, a_hi;
  {
    v8h x0 = *(const v8h*)(qbase + 8 * half);
    v8h x1 = *(const v8h*)(qbase + 16 + 8 * half);
    v8h x2 = *(const v8h*)(qbase + 32 + 8 * half);
    v8h x3 = *(const v8h*)(qbase + 48 + 8 * half);
    for (int j = 0; j < 8; ++j) {
      a_lo[j] = x0[j]; a_lo[j + 8] = x1[j];
      a_hi[j] = x2[j]; a_hi[j + 8] = x3[j];
    }
  }

  float m_run[8], l_run[8];
  v8f o[4];
  for (int r = 0; r < 8; ++r) { m_run[r] = NEGBIG; l_run[r] = 0.0f; }
  for (int t = 0; t < 4; ++t) o[t] = (v8f){};

  const size_t rowStride = (size_t)H_ * D_;   // floats between consecutive keys
  const float* kbase = kmat + ((size_t)b * L_ * H_ + h) * D_;
  const float* vbase = vmat + ((size_t)b * L_ * H_ + h) * D_;

  for (int kb = 0; kb < L_ / 32; ++kb) {
    int key0 = kb * 32;

    __syncthreads();   // previous iteration's kbuf/vbuf/pbuf reads complete

    // Stage K/V: one key row per lane, float4 loads, packed f16 LDS stores.
    {
      const float* kr = kbase + (size_t)(key0 + lane) * rowStride;
      const float* vr = vbase + (size_t)(key0 + lane) * rowStride;
      for (int d = 0; d < D_; d += 8) {
        v4f k0 = *(const v4f*)(kr + d);
        v4f k1 = *(const v4f*)(kr + d + 4);
        v4f v0 = *(const v4f*)(vr + d);
        v4f v1 = *(const v4f*)(vr + d + 4);
        v8h kh, vh;
        for (int j = 0; j < 4; ++j) {
          kh[j] = (_Float16)k0[j]; kh[j + 4] = (_Float16)k1[j];
          vh[j] = (_Float16)v0[j]; vh[j + 4] = (_Float16)v1[j];
        }
        *(v8h*)(&kbuf[lane][d]) = kh;
        *(v8h*)(&vbuf[lane][d]) = vh;
      }
    }
    // Prefetch next key block while this one computes.
    if (kb + 1 < L_ / 32) {
      __builtin_prefetch(kbase + (size_t)(key0 + 32 + lane) * rowStride, 0, 3);
      __builtin_prefetch(vbase + (size_t)(key0 + 32 + lane) * rowStride, 0, 3);
    }
    __syncthreads();

    // S tiles, one 16-key tile at a time (keeps one B-frag pair live).
    v8f c[2];
    for (int t16 = 0; t16 < 2; ++t16) {
      const _Float16* krow = &kbuf[t16 * 16 + nn][8 * half];
      v8h y0 = *(const v8h*)(krow);
      v8h y1 = *(const v8h*)(krow + 16);
      v8h y2 = *(const v8h*)(krow + 32);
      v8h y3 = *(const v8h*)(krow + 48);
      v16h blo, bhi;
      for (int j = 0; j < 8; ++j) {
        blo[j] = y0[j]; blo[j + 8] = y1[j];
        bhi[j] = y2[j]; bhi[j + 8] = y3[j];
      }
      v8f cc = (v8f){};
      cc = __builtin_amdgcn_wmma_f32_16x16x32_f16(false, a_lo, false, blo, (short)0, cc, false, false);
      cc = __builtin_amdgcn_wmma_f32_16x16x32_f16(false, a_hi, false, bhi, (short)0, cc, false, false);
      c[t16] = cc;
    }

    // attn_mask: masked keys -> -big before softmax
    if (amask[b * L_ + key0 + nn] == 0)      for (int r = 0; r < 8; ++r) c[0][r] = NEGBIG;
    if (amask[b * L_ + key0 + 16 + nn] == 0) for (int r = 0; r < 8; ++r) c[1][r] = NEGBIG;

    // Online softmax. C layout: lane nn = N, VGPR r = row (r + 8*half).
    // Butterfly step OUTER, rows INNER: 8 independent permutes per step.
    float rmax[8], rsum[8], corr[8], p0[8], p1[8];
    for (int r = 0; r < 8; ++r) rmax[r] = fmaxf(c[0][r], c[1][r]);
    for (int off = 1; off < 16; off <<= 1) {
      float t0[8];
      for (int r = 0; r < 8; ++r) t0[r] = __shfl_xor(rmax[r], off, 32);
      for (int r = 0; r < 8; ++r) rmax[r] = fmaxf(rmax[r], t0[r]);
    }
    for (int r = 0; r < 8; ++r) {
      float mn = fmaxf(m_run[r], rmax[r]);
      corr[r] = __expf(m_run[r] - mn);
      p0[r] = __expf(c[0][r] - mn);
      p1[r] = __expf(c[1][r] - mn);
      rsum[r] = p0[r] + p1[r];
      m_run[r] = mn;
    }
    for (int off = 1; off < 16; off <<= 1) {
      float t0[8];
      for (int r = 0; r < 8; ++r) t0[r] = __shfl_xor(rsum[r], off, 32);
      for (int r = 0; r < 8; ++r) rsum[r] += t0[r];
    }
    for (int r = 0; r < 8; ++r) l_run[r] = l_run[r] * corr[r] + rsum[r];
    for (int r = 0; r < 8; ++r)
      for (int t = 0; t < 4; ++t) o[t][r] *= corr[r];

    // Stage P (16 rows x 32 keys) through LDS: C-layout -> A-layout.
    for (int r = 0; r < 8; ++r) {
      pbuf[r + 8 * half][nn]      = (_Float16)p0[r];
      pbuf[r + 8 * half][16 + nn] = (_Float16)p1[r];
    }
    __syncthreads();
    v16h a_p;
    for (int j = 0; j < 16; ++j) {
      int kk = 16 * half + j;
      a_p[j] = pbuf[nn][kk];
    }

    // O += P(16x32) x V(32x64): 4 WMMAs over N tiles of 16.
    for (int t = 0; t < 4; ++t) {
      v16h bv;
      for (int j = 0; j < 16; ++j)
        bv[j] = vbuf[16 * half + j][t * 16 + nn];
      o[t] = __builtin_amdgcn_wmma_f32_16x16x32_f16(false, a_p, false, bv, (short)0, o[t], false, false);
    }
  }

  // Normalize and write: output is (B, U, H, D)
  for (int r = 0; r < 8; ++r) {
    int row = qt * 16 + r + 8 * half;
    if (row < U_) {
      float inv = 1.0f / l_run[r];
      for (int t = 0; t < 4; ++t) {
        out[(((size_t)(b * U_ + row) * H_ + h)) * D_ + t * 16 + nn] = o[t][r] * inv;
      }
    }
  }
}

// ---------------------------------------------------------------------------
extern "C" void kernel_launch(void* const* d_in, const int* in_sizes, int n_in,
                              void* d_out, int out_size, void* d_ws, size_t ws_size,
                              hipStream_t stream) {
  const float*     q    = (const float*)d_in[0];
  const float*     k    = (const float*)d_in[1];
  const float*     v    = (const float*)d_in[2];
  const int*       mask = (const int*)d_in[3];
  const long long* idx  = (const long long*)d_in[4];
  float* out = (float*)d_out;

  char* ws = (char*)d_ws;
  float*     Mws  = (float*)ws;                                   // B*H*L f32
  int*       Mtop = (int*)(ws + (size_t)B_ * H_ * L_ * 4);        // B*H*U i32
  _Float16*  Qr   = (_Float16*)(ws + (size_t)B_ * H_ * L_ * 4
                                   + (size_t)B_ * H_ * U_ * 4);   // B*H*48*64 f16

  int waves = B_ * H_ * L_;                       // 65536 waves, 8 per block
  probM_kernel<<<waves / 8, 256, 0, stream>>>(q, k, idx, Mws);
  topk_gather_kernel<<<B_ * H_, 256, 0, stream>>>(Mws, q, Mtop, Qr);
  probattn_kernel<<<B_ * H_ * QT_, 32, 0, stream>>>(Qr, k, v, mask, out);
}
